// ETransformerTrain_35021163331750
// MI455X (gfx1250) — matrix operand order repaired and checked by reading, not compiled, not run on hardware
//
#include <hip/hip_runtime.h>
#include <hip/hip_bf16.h>

// ---------------------------------------------------------------------------
// Problem constants (from reference)
// ---------------------------------------------------------------------------
#define E_EDGES   32768
#define S_SEQ     16
#define H_DIM     128
#define D_IN      128
#define EDGE_IN   16
#define N_SRC     20000
#define N_DST     5000
#define HEADS     8
#define DH        16
#define LAYERS    2
#define NUM_CLASS 10
#define TOKENS    (E_EDGES * S_SEQ)      // 524288

typedef __attribute__((ext_vector_type(16))) __bf16 v16bf;
typedef __attribute__((ext_vector_type(8)))  __bf16 v8bf;
typedef __attribute__((ext_vector_type(8)))  float  v8f;

__device__ __forceinline__ __bf16 f2bf(float f) {
    unsigned u = __builtin_bit_cast(unsigned, f);
    u += 0x7FFFu + ((u >> 16) & 1u);               // round-to-nearest-even
    unsigned short s = (unsigned short)(u >> 16);
    return __builtin_bit_cast(__bf16, s);
}
__device__ __forceinline__ float bf2f(__bf16 h) {
    unsigned short s = __builtin_bit_cast(unsigned short, h);
    unsigned u = ((unsigned)s) << 16;
    return __builtin_bit_cast(float, u);
}
__device__ __forceinline__ v8bf zero_v8bf() {
    __bf16 b = __builtin_bit_cast(__bf16, (unsigned short)0);
    v8bf z;
    #pragma unroll
    for (int i = 0; i < 8; ++i) z[i] = b;
    return z;
}
// low 32 bits of a generic __shared__ address = wave-relative LDS byte offset
__device__ __forceinline__ unsigned lds_off(const void* p) {
    return (unsigned)(uintptr_t)p;
}
__device__ __forceinline__ void async_load_b128_to_lds(const void* gaddr, unsigned ldsaddr) {
    asm volatile("global_load_async_to_lds_b128 %0, %1, off"
                 :: "v"(ldsaddr), "v"(gaddr) : "memory");
}
__device__ __forceinline__ void wait_asynccnt0() {
    asm volatile("s_wait_asynccnt 0x0" ::: "memory");
}

// ---------------------------------------------------------------------------
// WMMA GEMM:  C[M,N] = act(A[M,K] @ W + bias[N])
//   A  : bf16 row-major [M,K]            (k-contiguous)
//   WT : bf16 pre-transposed [N,K]       (k-contiguous)
// Tile 64x128, BK=32; 256 threads = 8 waves; each wave owns one 16-row strip
// and four 16x16 accumulators.  Double-buffered LDS; staging for the hot
// (MCHECK=0) path uses CDNA5 async Global->LDS b128 loads tracked by ASYNCcnt,
// overlapped with WMMA on the other buffer.  One barrier per K step.
// ---------------------------------------------------------------------------
#define BM 64
#define BN 128
#define BK 32
#define BKP 40

template<int OUT_BF16, int RELU, int MCHECK>
__global__ __launch_bounds__(256)
void gemm_wmma(const __bf16* __restrict__ A, const __bf16* __restrict__ WT,
               const float* __restrict__ bias, void* __restrict__ Cout,
               int M, int N, int K)
{
    __shared__ __align__(16) __bf16 As [2][BM][BKP];
    __shared__ __align__(16) __bf16 BsT[2][BN][BKP];

    const int tid  = threadIdx.x;
    const int wave = tid >> 5;
    const int lane = tid & 31;
    const int wm16 = (wave & 3) * 16;        // wave's M tile inside block
    const int wn   = (wave >> 2) * 64;       // wave's N quad base (0 or 64)
    const int m0   = blockIdx.x * BM;
    const int n0   = blockIdx.y * BN;

    v8f acc[4];
    #pragma unroll
    for (int t = 0; t < 4; ++t) acc[t] = (v8f){0.f,0.f,0.f,0.f,0.f,0.f,0.f,0.f};

    // per-thread staging coordinates (16B granules)
    const int ar  = tid >> 2;            // A row 0..63
    const int ac8 = (tid & 3) << 3;      // A col 0,8,16,24

    auto stage = [&](int k0, int buf) {
        if (MCHECK) {
            // synchronous predicated staging (OOB rows -> zeros)
            v8bf v;
            if ((m0 + ar) < M) v = *(const v8bf*)&A[(size_t)(m0 + ar) * K + k0 + ac8];
            else               v = zero_v8bf();
            *(v8bf*)&As[buf][ar][ac8] = v;
            #pragma unroll
            for (int t = 0; t < 2; ++t) {
                int idx = tid + t * 256;
                int n  = idx >> 2;
                int k8 = (idx & 3) << 3;
                *(v8bf*)&BsT[buf][n][k8] = *(const v8bf*)&WT[(size_t)(n0 + n) * K + k0 + k8];
            }
        } else {
            // CDNA5 async Global->LDS (ASYNCcnt), no VGPR round-trip
            async_load_b128_to_lds(&A[(size_t)(m0 + ar) * K + k0 + ac8],
                                   lds_off(&As[buf][ar][ac8]));
            #pragma unroll
            for (int t = 0; t < 2; ++t) {
                int idx = tid + t * 256;
                int n  = idx >> 2;
                int k8 = (idx & 3) << 3;
                async_load_b128_to_lds(&WT[(size_t)(n0 + n) * K + k0 + k8],
                                       lds_off(&BsT[buf][n][k8]));
            }
        }
    };

    stage(0, 0);
    int buf = 0;
    for (int k0 = 0; k0 < K; k0 += BK) {
        if (!MCHECK) wait_asynccnt0();   // own async loads into `buf` complete
        __syncthreads();                  // everyone's loads into `buf` complete
        if (k0 + BK < K) stage(k0 + BK, buf ^ 1);   // prefetch next tile

        // ---- A fragment: two contiguous 16B halves -> ds_load_b128 x2 ----
        const int am  = lane & 15;
        const int kbA = (lane >> 4) << 3;    // 0 or 8
        const __bf16* ap = &As[buf][wm16 + am][0];
        v8bf alo = *(const v8bf*)(ap + kbA);
        v8bf ahi = *(const v8bf*)(ap + 16 + kbA);
        v16bf af = __builtin_shufflevector(alo, ahi,
                    0,1,2,3,4,5,6,7,8,9,10,11,12,13,14,15);

        // ---- B fragments: two 8-K halves per lane in BsT (same k order as A) ----
        const int bn  = lane & 15;
        const int kbB = (lane >> 4) << 3;    // 0 or 8
        #pragma unroll
        for (int t = 0; t < 4; ++t) {
            const __bf16* bp = &BsT[buf][wn + t * 16 + bn][0];
            v8bf blo = *(const v8bf*)(bp + kbB);
            v8bf bhi = *(const v8bf*)(bp + 16 + kbB);
            v16bf bfr = __builtin_shufflevector(blo, bhi,
                         0,1,2,3,4,5,6,7,8,9,10,11,12,13,14,15);
            acc[t] = __builtin_amdgcn_wmma_f32_16x16x32_bf16(false, af, false, bfr,
                                                             (short)0, acc[t], false, false);
        }
        buf ^= 1;
    }

    // ---- epilogue: C layout VGPR r -> M = r + 8*(lane>=16), N = lane&15 ----
    const int rbase = m0 + wm16 + ((lane >> 4) << 3);
    #pragma unroll
    for (int t = 0; t < 4; ++t) {
        const int col = n0 + wn + t * 16 + (lane & 15);
        const float bb = bias[col];
        #pragma unroll
        for (int r = 0; r < 8; ++r) {
            int row = rbase + r;
            if (!MCHECK || row < M) {
                float v = acc[t][r] + bb;
                if (RELU) v = fmaxf(v, 0.f);
                if (OUT_BF16) ((__bf16*)Cout)[(size_t)row * N + col] = f2bf(v);
                else          ((float*) Cout)[(size_t)row * N + col] = v;
            }
        }
    }
}

// ---------------------------------------------------------------------------
// Weight pre-pass: WT[n*K + k] = bf16(W[k*N + n])   (tiny, once per launch)
// ---------------------------------------------------------------------------
__global__ void wtrans_kernel(const float* __restrict__ W, __bf16* __restrict__ WT,
                              int K, int N)
{
    int idx = blockIdx.x * blockDim.x + threadIdx.x;
    if (idx >= K * N) return;
    int k = idx / N, n = idx % N;
    WT[(size_t)n * K + k] = f2bf(W[idx]);
}

// ---------------------------------------------------------------------------
// Edge embedding: relu([edge_features, seq_time] @ W_ein + b_ein), K=16
// Dual-writes f32 (residual path) + bf16 (GEMM A path).
// ---------------------------------------------------------------------------
__global__ __launch_bounds__(128)
void embed_kernel(const float* __restrict__ ef, const float* __restrict__ st,
                  const float* __restrict__ W, const float* __restrict__ b,
                  float* __restrict__ out, __bf16* __restrict__ outbf)
{
    __shared__ float xs[EDGE_IN];
    const int tok = blockIdx.x;
    const int t = threadIdx.x;
    if (t < EDGE_IN) xs[t] = (t < EDGE_IN - 1) ? ef[(size_t)tok * (EDGE_IN - 1) + t]
                                               : st[tok];
    __syncthreads();
    float acc = b[t];
    #pragma unroll
    for (int k = 0; k < EDGE_IN; ++k) acc += xs[k] * W[k * H_DIM + t];
    acc = fmaxf(acc, 0.f);
    out  [(size_t)tok * H_DIM + t] = acc;
    outbf[(size_t)tok * H_DIM + t] = f2bf(acc);
}

// ---------------------------------------------------------------------------
// Per-edge multi-head attention (S=16, 8 heads x DH=16), qkv bf16 -> ctx bf16.
// ---------------------------------------------------------------------------
__global__ __launch_bounds__(128)
void attn_kernel(const __bf16* __restrict__ qkv, const int* __restrict__ elen,
                 __bf16* __restrict__ ctx)
{
    __shared__ float ks[S_SEQ][H_DIM];
    __shared__ float vs[S_SEQ][H_DIM];
    const int e = blockIdx.x;
    const __bf16* base = qkv + (size_t)e * S_SEQ * (3 * H_DIM);

    for (int idx = threadIdx.x; idx < S_SEQ * H_DIM; idx += 128) {
        int s = idx >> 7, c = idx & 127;
        ks[s][c] = bf2f(base[s * (3 * H_DIM) + H_DIM + c]);
        vs[s][c] = bf2f(base[s * (3 * H_DIM) + 2 * H_DIM + c]);
    }
    __syncthreads();

    const int i  = threadIdx.x & 15;
    const int hd = threadIdx.x >> 4;
    const int len = elen[e];

    float q[DH];
    #pragma unroll
    for (int d = 0; d < DH; ++d) q[d] = bf2f(base[i * (3 * H_DIM) + hd * DH + d]);

    float sc[S_SEQ];
    float mx = -1e30f;
    #pragma unroll
    for (int j = 0; j < S_SEQ; ++j) {
        float a;
        if (j < len) {
            a = 0.f;
            #pragma unroll
            for (int d = 0; d < DH; ++d) a += q[d] * ks[j][hd * DH + d];
            a *= 0.25f;                       // 1/sqrt(16)
        } else {
            a = -1e9f;
        }
        sc[j] = a;
        mx = fmaxf(mx, a);
    }
    float ssum = 0.f;
    #pragma unroll
    for (int j = 0; j < S_SEQ; ++j) { sc[j] = __expf(sc[j] - mx); ssum += sc[j]; }
    const float inv = 1.f / ssum;

    float o[DH];
    #pragma unroll
    for (int d = 0; d < DH; ++d) o[d] = 0.f;
    #pragma unroll
    for (int j = 0; j < S_SEQ; ++j) {
        float w = sc[j] * inv;
        #pragma unroll
        for (int d = 0; d < DH; ++d) o[d] += w * vs[j][hd * DH + d];
    }
    __bf16* op = ctx + ((size_t)e * S_SEQ + i) * H_DIM + hd * DH;
    #pragma unroll
    for (int d = 0; d < DH; ++d) op[d] = f2bf(o[d]);
}

// ---------------------------------------------------------------------------
// LayerNorm over H=128 of (X + R); dual-writes f32 + bf16.
// ---------------------------------------------------------------------------
__global__ __launch_bounds__(256)
void ln_kernel(const float* __restrict__ X, const float* __restrict__ R,
               const float* __restrict__ g, const float* __restrict__ b,
               float* __restrict__ Out, __bf16* __restrict__ Outbf, int rows)
{
    const int wave = threadIdx.x >> 5;
    const int lane = threadIdx.x & 31;
    const int row = blockIdx.x * 8 + wave;
    if (row >= rows) return;

    float x[4];
    float s = 0.f;
    #pragma unroll
    for (int i = 0; i < 4; ++i) {
        int c = lane + i * 32;
        x[i] = X[(size_t)row * H_DIM + c] + R[(size_t)row * H_DIM + c];
        s += x[i];
    }
    #pragma unroll
    for (int off = 16; off >= 1; off >>= 1) s += __shfl_xor(s, off, 32);
    const float mu = s * (1.f / H_DIM);

    float v = 0.f;
    #pragma unroll
    for (int i = 0; i < 4; ++i) { float d = x[i] - mu; v += d * d; }
    #pragma unroll
    for (int off = 16; off >= 1; off >>= 1) v += __shfl_xor(v, off, 32);
    const float rs = rsqrtf(v * (1.f / H_DIM) + 1e-5f);

    #pragma unroll
    for (int i = 0; i < 4; ++i) {
        int c = lane + i * 32;
        float y = (x[i] - mu) * rs * g[c] + b[c];
        Out  [(size_t)row * H_DIM + c] = y;
        Outbf[(size_t)row * H_DIM + c] = f2bf(y);
    }
}

// ---------------------------------------------------------------------------
// Small elementwise / gather / scatter kernels
// ---------------------------------------------------------------------------
__global__ void pool_kernel(const float* __restrict__ e, const int* __restrict__ elen,
                            float* __restrict__ out, int total)
{
    int idx = blockIdx.x * blockDim.x + threadIdx.x;
    if (idx >= total) return;
    int ed = idx >> 7, c = idx & 127;
    int len = elen[ed];
    float s = 0.f;
    for (int j = 0; j < len; ++j) s += e[((size_t)ed * S_SEQ + j) * H_DIM + c];
    out[idx] = s / (float)len;
}

// msg A matrix (bf16): [h[src], pool]
__global__ void msg_concat_kernel(const float* __restrict__ nf, const int* __restrict__ src,
                                  const float* __restrict__ pool, __bf16* __restrict__ out, int total)
{
    int idx = blockIdx.x * blockDim.x + threadIdx.x;
    if (idx >= total) return;
    int e = idx >> 8, c = idx & 255;
    float v = (c < 128) ? nf[(size_t)src[e] * D_IN + c]
                        : pool[(size_t)e * H_DIM + (c - 128)];
    out[idx] = f2bf(v);
}

// gather self_h: dual f32 + bf16
__global__ void gather_kernel(const float* __restrict__ nf, const int* __restrict__ nid,
                              float* __restrict__ out, __bf16* __restrict__ outbf, int total)
{
    int idx = blockIdx.x * blockDim.x + threadIdx.x;
    if (idx >= total) return;
    int n = idx >> 7, c = idx & 127;
    float v = nf[(size_t)nid[n] * D_IN + c];
    out[idx] = v;
    outbf[idx] = f2bf(v);
}

__global__ void fill_kernel(float* __restrict__ p, float v, int n)
{
    int idx = blockIdx.x * blockDim.x + threadIdx.x;
    if (idx < n) p[idx] = v;
}

__global__ void scatter_kernel(const float* __restrict__ m, const int* __restrict__ dst,
                               float* __restrict__ agg, int total)
{
    int idx = blockIdx.x * blockDim.x + threadIdx.x;
    if (idx >= total) return;
    int e = idx >> 7, c = idx & 127;
    atomicAdd(&agg[(size_t)dst[e] * H_DIM + c], m[idx]);
}

// node A matrix (bf16): [self_h, (agg - self_tmp)*norm]
__global__ void node_concat_kernel(const float* __restrict__ selfH, const float* __restrict__ agg,
                                   const float* __restrict__ selfTmp, const float* __restrict__ norm,
                                   __bf16* __restrict__ out, int total)
{
    int idx = blockIdx.x * blockDim.x + threadIdx.x;
    if (idx >= total) return;
    int n = idx >> 8, c = idx & 255;
    float v = (c < 128) ? selfH[(size_t)n * 128 + c]
                        : (agg[(size_t)n * 128 + (c - 128)] - selfTmp[(size_t)n * 128 + (c - 128)]) * norm[n];
    out[idx] = f2bf(v);
}

__global__ void fc_kernel(const float* __restrict__ z, const float* __restrict__ W,
                          const float* __restrict__ b, float* __restrict__ out, int total)
{
    int idx = blockIdx.x * blockDim.x + threadIdx.x;
    if (idx >= total) return;
    int n = idx / NUM_CLASS, c = idx % NUM_CLASS;
    float acc = b[c];
    for (int k = 0; k < H_DIM; ++k) acc += z[(size_t)n * H_DIM + k] * W[k * NUM_CLASS + c];
    out[idx] = acc;
}

// ---------------------------------------------------------------------------
// Host launcher
// ---------------------------------------------------------------------------
extern "C" void kernel_launch(void* const* d_in, const int* in_sizes, int n_in,
                              void* d_out, int out_size, void* d_ws, size_t ws_size,
                              hipStream_t stream)
{
    // ---- inputs (setup_inputs dict order) ----
    const float* node_features = (const float*)d_in[0];
    const float* edge_features = (const float*)d_in[1];
    const float* seq_times     = (const float*)d_in[2];
    const int*   e_len         = (const int*)d_in[3];
    // d_in[4] e_mask (bool) -- recomputed from e_len on device
    const int*   src_ids       = (const int*)d_in[5];
    const int*   dst_ids       = (const int*)d_in[6];
    const int*   layer_nid     = (const int*)d_in[7];
    const float* subg_norm     = (const float*)d_in[8];
    const float* W_ein   = (const float*)d_in[9];
    const float* b_ein   = (const float*)d_in[10];
    const float* Wqkv    = (const float*)d_in[11];
    const float* bqkv    = (const float*)d_in[12];
    const float* Wo      = (const float*)d_in[13];
    const float* bo      = (const float*)d_in[14];
    const float* ln1g    = (const float*)d_in[15];
    const float* ln1b    = (const float*)d_in[16];
    const float* W1      = (const float*)d_in[17];
    const float* b1      = (const float*)d_in[18];
    const float* W2      = (const float*)d_in[19];
    const float* b2      = (const float*)d_in[20];
    const float* ln2g    = (const float*)d_in[21];
    const float* ln2b    = (const float*)d_in[22];
    const float* W_eout  = (const float*)d_in[23];
    const float* b_eout  = (const float*)d_in[24];
    const float* W_node  = (const float*)d_in[25];
    const float* b_node  = (const float*)d_in[26];
    const float* W_fc    = (const float*)d_in[27];
    const float* b_fc    = (const float*)d_in[28];
    float* out = (float*)d_out;

    // ---- workspace layout ----
    char* w = (char*)d_ws;
    size_t off = 0;
    auto take = [&](size_t bytes) { void* p = w + off; off += (bytes + 255) & ~(size_t)255; return p; };
    float*  bufE    = (float*) take((size_t)TOKENS * H_DIM * 4);        // f32 activations
    __bf16* bufEbf  = (__bf16*)take((size_t)TOKENS * H_DIM * 2);        // bf16 mirror (GEMM A)
    __bf16* bufQKV  = (__bf16*)take((size_t)TOKENS * 3 * H_DIM * 2);    // qkv bf16
    __bf16* bufTbf  = (__bf16*)take((size_t)TOKENS * H_DIM * 2);        // ctx / ff1 (bf16)
    float*  bufT2   = (float*) take((size_t)TOKENS * H_DIM * 4);        // gemm f32 out
    float*  bufPool = (float*) take((size_t)E_EDGES * H_DIM * 4);
    __bf16* msgA    = (__bf16*)take((size_t)E_EDGES * 256 * 2);
    float*  bufM    = (float*) take((size_t)E_EDGES * H_DIM * 4);
    float*  selfHf  = (float*) take((size_t)N_DST * 128 * 4);
    __bf16* selfHbf = (__bf16*)take((size_t)N_DST * 128 * 2);
    float*  selfTmp = (float*) take((size_t)N_DST * 128 * 4);
    float*  agg     = (float*) take((size_t)N_DST * 128 * 4);
    __bf16* nodeA   = (__bf16*)take((size_t)N_DST * 256 * 2);
    float*  zbuf    = (float*) take((size_t)N_DST * 128 * 4);
    // pre-transposed bf16 weights WT[n][k]
    __bf16* WqkvT   = (__bf16*)take((size_t)LAYERS * 384 * 128 * 2);
    __bf16* WoT     = (__bf16*)take((size_t)LAYERS * 128 * 128 * 2);
    __bf16* W1T     = (__bf16*)take((size_t)LAYERS * 128 * 128 * 2);
    __bf16* W2T     = (__bf16*)take((size_t)LAYERS * 128 * 128 * 2);
    __bf16* WeoutT  = (__bf16*)take((size_t)128 * 256 * 2);             // [n=128][k=256]
    __bf16* WeoutT128 = (__bf16*)take((size_t)128 * 128 * 2);           // first 128 rows
    __bf16* WnodeT  = (__bf16*)take((size_t)128 * 256 * 2);
    (void)ws_size; (void)n_in; (void)in_sizes; (void)out_size;

    auto gemm = [&](const __bf16* A, const __bf16* WT, const float* bias, void* C,
                    int M, int N, int K, bool relu, bool outbf) {
        dim3 g((M + BM - 1) / BM, N / BN), blk(256);
        bool mchk = (M % BM) != 0;
        if (outbf) {
            if (relu) { if (mchk) gemm_wmma<1,1,1><<<g,blk,0,stream>>>(A,WT,bias,C,M,N,K);
                        else      gemm_wmma<1,1,0><<<g,blk,0,stream>>>(A,WT,bias,C,M,N,K); }
            else      { if (mchk) gemm_wmma<1,0,1><<<g,blk,0,stream>>>(A,WT,bias,C,M,N,K);
                        else      gemm_wmma<1,0,0><<<g,blk,0,stream>>>(A,WT,bias,C,M,N,K); }
        } else {
            if (relu) { if (mchk) gemm_wmma<0,1,1><<<g,blk,0,stream>>>(A,WT,bias,C,M,N,K);
                        else      gemm_wmma<0,1,0><<<g,blk,0,stream>>>(A,WT,bias,C,M,N,K); }
            else      { if (mchk) gemm_wmma<0,0,1><<<g,blk,0,stream>>>(A,WT,bias,C,M,N,K);
                        else      gemm_wmma<0,0,0><<<g,blk,0,stream>>>(A,WT,bias,C,M,N,K); }
        }
    };
    auto wtrans = [&](const float* W, __bf16* WT, int K, int N) {
        int total = K * N;
        wtrans_kernel<<<(total + 255) / 256, 256, 0, stream>>>(W, WT, K, N);
    };

    // 0) weight pre-pass (tiny)
    for (int l = 0; l < LAYERS; ++l) {
        wtrans(Wqkv + (size_t)l * 128 * 384, WqkvT + (size_t)l * 384 * 128, 128, 384);
        wtrans(Wo   + (size_t)l * 128 * 128, WoT   + (size_t)l * 128 * 128, 128, 128);
        wtrans(W1   + (size_t)l * 128 * 128, W1T   + (size_t)l * 128 * 128, 128, 128);
        wtrans(W2   + (size_t)l * 128 * 128, W2T   + (size_t)l * 128 * 128, 128, 128);
    }
    wtrans(W_eout, WeoutT,    256, 128);
    wtrans(W_eout, WeoutT128, 128, 128);   // only first 128 rows (zero-concat half)
    wtrans(W_node, WnodeT,    256, 128);

    // 1) edge token embedding -> bufE (f32) + bufEbf (bf16)
    embed_kernel<<<TOKENS, 128, 0, stream>>>(edge_features, seq_times, W_ein, b_ein, bufE, bufEbf);

    // 2) transformer encoder, 2 layers, post-norm
    for (int l = 0; l < LAYERS; ++l) {
        const __bf16* WqkvT_l = WqkvT + (size_t)l * 384 * 128;
        const __bf16* WoT_l   = WoT   + (size_t)l * 128 * 128;
        const __bf16* W1T_l   = W1T   + (size_t)l * 128 * 128;
        const __bf16* W2T_l   = W2T   + (size_t)l * 128 * 128;

        gemm(bufEbf, WqkvT_l, bqkv + (size_t)l * 384, bufQKV, TOKENS, 384, 128, false, true);
        attn_kernel<<<E_EDGES, 128, 0, stream>>>(bufQKV, e_len, bufTbf);
        gemm(bufTbf, WoT_l, bo + (size_t)l * 128, bufT2, TOKENS, 128, 128, false, false);
        ln_kernel<<<TOKENS / 8, 256, 0, stream>>>(bufT2, bufE,
                                                  ln1g + (size_t)l * 128, ln1b + (size_t)l * 128,
                                                  bufE, bufEbf, TOKENS);
        gemm(bufEbf, W1T_l, b1 + (size_t)l * 128, bufTbf, TOKENS, 128, 128, true, true);
        gemm(bufTbf, W2T_l, b2 + (size_t)l * 128, bufT2, TOKENS, 128, 128, false, false);
        ln_kernel<<<TOKENS / 8, 256, 0, stream>>>(bufT2, bufE,
                                                  ln2g + (size_t)l * 128, ln2b + (size_t)l * 128,
                                                  bufE, bufEbf, TOKENS);
    }

    // 3) masked mean pool over sequence -> bufPool [E,128] (f32)
    {
        int total = E_EDGES * H_DIM;
        pool_kernel<<<(total + 255) / 256, 256, 0, stream>>>(bufE, e_len, bufPool, total);
    }
    // 4) messages: relu([h[src], pool] @ W_eout + b_eout)
    {
        int total = E_EDGES * 256;
        msg_concat_kernel<<<(total + 255) / 256, 256, 0, stream>>>(node_features, src_ids, bufPool, msgA, total);
        gemm(msgA, WeoutT, b_eout, bufM, E_EDGES, 128, 256, true, false);
    }
    // 5) self_h and self_h_tmp = self_h @ W_eout[0:128,:] + b_eout
    {
        int total = N_DST * 128;
        gather_kernel<<<(total + 255) / 256, 256, 0, stream>>>(node_features, layer_nid, selfHf, selfHbf, total);
        gemm(selfHbf, WeoutT128, b_eout, selfTmp, N_DST, 128, 128, false, false);
    }
    // 6) segment sum of messages into agg
    {
        int total = N_DST * 128;
        fill_kernel<<<(total + 255) / 256, 256, 0, stream>>>(agg, 0.f, total);
        int total2 = E_EDGES * H_DIM;
        scatter_kernel<<<(total2 + 255) / 256, 256, 0, stream>>>(bufM, dst_ids, agg, total2);
    }
    // 7) node update: z = relu([self_h, (agg - self_tmp)*norm] @ W_node + b_node)
    {
        int total = N_DST * 256;
        node_concat_kernel<<<(total + 255) / 256, 256, 0, stream>>>(selfHf, agg, selfTmp, subg_norm, nodeA, total);
        gemm(nodeA, WnodeT, b_node, zbuf, N_DST, 128, 256, true, false);
    }
    // 8) classifier
    {
        int total = N_DST * NUM_CLASS;
        fc_kernel<<<(total + 255) / 256, 256, 0, stream>>>(zbuf, W_fc, b_fc, out, total);
    }
}
